// RModel_29257317221019
// MI455X (gfx1250) — matrix operand order repaired and checked by reading, not compiled, not run on hardware
//
#include <hip/hip_runtime.h>

typedef __attribute__((ext_vector_type(16))) _Float16 v16h;
typedef __attribute__((ext_vector_type(8)))  float    v8f;
typedef __attribute__((ext_vector_type(8)))  int      v8i;
typedef __attribute__((ext_vector_type(2)))  _Float16 v2h;

#define WPB   8                    // waves per block (256 threads, wave32)
#define ITERS 8                    // 32-point chunks per wave
#define PPB   (WPB * ITERS * 32)   // 2048 points per block
#define INV2PI 0.15915494309189535f

// v_cos_f32 takes turns (cos(2*pi*t)); all args below are pre-scaled to turns.
__device__ __forceinline__ float cos_turns(float t) {
  return __builtin_amdgcn_cosf(t);
}

// K index of A-fragment element pair {2v,2v+1} for 16-bit A 16x32 (contiguous: k = 2v + 16*hlf, matches B)
__device__ __forceinline__ int a_k(int v, int hlf) {
  return 2 * v + 16 * hlf;
}

__device__ __forceinline__ int pack2h(float a, float b) {
  v2h p; p[0] = (_Float16)a; p[1] = (_Float16)b;   // -> v_cvt_pk_f16_f32
  return __builtin_bit_cast(int, p);
}

// VALU half-wave swap: lane i gets src from lane (i+16)%32 (v_permlanex16_b32, identity sel).
__device__ __forceinline__ int permx16(int v) {
  return __builtin_amdgcn_permlanex16(v, v, 0x76543210, 0xFEDCBA98, false, false);
}

// A fragment of W^T from TRANSPOSED LDS image wt[ch][k] (ld = 32).
// Row m = output channel chBase+m, K = input channel; pairs (k,k+1) contiguous -> b32 loads.
__device__ __forceinline__ v16h load_aT_frag(const _Float16* wt, int chBase, int lane) {
  const int hlf = lane >> 4, m = lane & 15;
  const int ch  = chBase + m;
  v16h a;
#pragma unroll
  for (int v = 0; v < 8; ++v) {
    const int k = a_k(v, hlf);
    a[2 * v]     = wt[ch * 32 + k];
    a[2 * v + 1] = wt[ch * 32 + k + 1];
  }
  return a;
}

// Turn two post-activation D^T tiles (f32, C-layout: tile u holds channels 16u + r + 8*srcHalf,
// point = lane&15) into the next layer's B fragment (k = channel, n = point).
// Dest lane l, element v: channels (2v+16*hlf, +1) -> tile hlf, VGPR pair 2v%8,
// source lane (lane&15) for v<4 else (lane&15)|16. All cross-lane moves are half-swaps.
__device__ __forceinline__ v16h make_b_frag(const float* c0, const float* c1, int hlf) {
  v8i b;
#pragma unroll
  for (int j = 0; j < 4; ++j) {
    const int r0  = 2 * j;
    const int pk0 = pack2h(c0[r0], c0[r0 + 1]);
    const int pk1 = pack2h(c1[r0], c1[r0 + 1]);
    const int x0  = permx16(pk0);
    const int x1  = permx16(pk1);
    b[j]     = hlf ? x1 : pk0;   // v = j:   lower half own pk0, upper half pk1 from lane-16
    b[j + 4] = hlf ? pk1 : x0;   // v = j+4: lower half pk0 from lane+16, upper half own pk1
  }
  return __builtin_bit_cast(v16h, b);
}

__global__ __launch_bounds__(256)
void deform_sample_kernel(const float* __restrict__ x,
                          const float* __restrict__ W1, const float* __restrict__ b1,
                          const float* __restrict__ W2, const float* __restrict__ b2,
                          const float* __restrict__ W3, const float* __restrict__ b3,
                          const float* __restrict__ Wf, const float* __restrict__ bf,
                          const float* __restrict__ vol,
                          float* __restrict__ out, long N)
{
  __shared__ __align__(16) float sW1t[32 * 4];   // [k][{w0,w1,w2,b1}] f32, pre-scaled by 1/2pi
  __shared__ _Float16 sW2t[32 * 32];             // transposed: [ch][k] f16
  __shared__ _Float16 sW3t[32 * 32];             // transposed: [ch][k] f16
  __shared__ _Float16 sWft[16 * 32];             // transposed: [m][k] f16, rows 3..15 zero
  __shared__ float    sB2[32], sB3[32];

  const int tid  = threadIdx.x;
  const int wid  = tid >> 5;
  const int lane = tid & 31;
  const int hlf  = lane >> 4;
  const int nl   = lane & 15;

  // ---- stage weights into LDS ----
  for (int i = tid; i < 16 * 32; i += 256) sWft[i] = (_Float16)0.0f;
  __syncthreads();
  for (int i = tid; i < 96; i += 256)   sW1t[(i % 32) * 4 + (i / 32)] = W1[i] * INV2PI;
  for (int i = tid; i < 32; i += 256)   { sW1t[i * 4 + 3] = b1[i] * INV2PI; sB2[i] = b2[i]; sB3[i] = b3[i]; }
  for (int i = tid; i < 1024; i += 256) {
    const int k = i >> 5, n = i & 31;                 // source [k][n] -> dest [n][k]
    sW2t[n * 32 + k] = (_Float16)W2[i];
    sW3t[n * 32 + k] = (_Float16)W3[i];
  }
  for (int i = tid; i < 96; i += 256)   sWft[(i % 3) * 32 + (i / 3)] = (_Float16)Wf[i];
  __syncthreads();

  // ---- persistent transposed-weight A fragments + per-lane biases (pre-scaled to turns) ----
  v16h A2[2], A3[2], AF;
  A2[0] = load_aT_frag(sW2t, 0, lane);  A2[1] = load_aT_frag(sW2t, 16, lane);
  A3[0] = load_aT_frag(sW3t, 0, lane);  A3[1] = load_aT_frag(sW3t, 16, lane);
  AF    = load_aT_frag(sWft, 0, lane);
  float bias2[16], bias3[16];
#pragma unroll
  for (int u = 0; u < 2; ++u)
#pragma unroll
    for (int r = 0; r < 8; ++r) {
      bias2[u * 8 + r] = sB2[16 * u + r + 8 * hlf] * INV2PI;
      bias3[u * 8 + r] = sB3[16 * u + r + 8 * hlf] * INV2PI;
    }
  const float bf0 = bf[0], bf1 = bf[1], bf2 = bf[2];

  const long blockBase = (long)blockIdx.x * PPB;

  for (int it = 0; it < ITERS; ++it) {
    const long chunk = blockBase + (long)wid * (ITERS * 32) + (long)it * 32;
    float dsub[2][3];

#pragma unroll
    for (int sub = 0; sub < 2; ++sub) {
      // this lane owns point column n = nl of the 16-point tile
      const long pm = chunk + sub * 16 + nl;
      float cx = 0.f, cy = 0.f, cz = 0.f;
      if (pm < N) {
        const float* xp = x + 4 * pm;
        cx = xp[0]; cy = xp[1]; cz = xp[2];
      }

      // ---- layer 1 (3->32) in f32 VALU (weights pre-scaled: arg already in turns) ----
      v8i b1i;
#pragma unroll
      for (int v = 0; v < 8; ++v) {
        const int k = 2 * v + 16 * hlf;
        const float4 w0 = *(const float4*)&sW1t[k * 4];
        const float4 w1 = *(const float4*)&sW1t[(k + 1) * 4];
        const float h0 = cos_turns(cx * w0.x + cy * w0.y + cz * w0.z + w0.w);
        const float h1 = cos_turns(cx * w1.x + cy * w1.y + cz * w1.z + w1.w);
        b1i[v] = pack2h(h0, h1);
      }
      v16h B = __builtin_bit_cast(v16h, b1i);

      // ---- layer 2: issue both WMMAs, then one fma + cos per element ----
      float c0[8], c1[8];
      {
        v8f t0 = {}, t1 = {};
        t0 = __builtin_amdgcn_wmma_f32_16x16x32_f16(false, A2[0], false, B,
                                                    (short)0, t0, false, false);
        t1 = __builtin_amdgcn_wmma_f32_16x16x32_f16(false, A2[1], false, B,
                                                    (short)0, t1, false, false);
#pragma unroll
        for (int r = 0; r < 8; ++r) c0[r] = cos_turns(fmaf(t0[r], INV2PI, bias2[r]));
#pragma unroll
        for (int r = 0; r < 8; ++r) c1[r] = cos_turns(fmaf(t1[r], INV2PI, bias2[8 + r]));
      }
      B = make_b_frag(c0, c1, hlf);

      // ---- layer 3 ----
      {
        v8f t0 = {}, t1 = {};
        t0 = __builtin_amdgcn_wmma_f32_16x16x32_f16(false, A3[0], false, B,
                                                    (short)0, t0, false, false);
        t1 = __builtin_amdgcn_wmma_f32_16x16x32_f16(false, A3[1], false, B,
                                                    (short)0, t1, false, false);
#pragma unroll
        for (int r = 0; r < 8; ++r) c0[r] = cos_turns(fmaf(t0[r], INV2PI, bias3[r]));
#pragma unroll
        for (int r = 0; r < 8; ++r) c1[r] = cos_turns(fmaf(t1[r], INV2PI, bias3[8 + r]));
      }
      B = make_b_frag(c0, c1, hlf);

      // ---- final 32->3: D^T rows 0..2 valid on lanes 0..15 (point = lane) ----
      v8f d = {};
      d = __builtin_amdgcn_wmma_f32_16x16x32_f16(false, AF, false, B,
                                                 (short)0, d, false, false);
      dsub[sub][0] = d[0]; dsub[sub][1] = d[1]; dsub[sub][2] = d[2];
    }

    // route deltas to sampling lanes: s<16 -> sub0 own lane; s>=16 -> sub1 from lane s-16
    float dd[3];
#pragma unroll
    for (int j = 0; j < 3; ++j) {
      const int sw = permx16(__builtin_bit_cast(int, dsub[1][j]));
      dd[j] = (lane < 16) ? dsub[0][j] : __builtin_bit_cast(float, sw);
    }

    // ---- trilinear sample: all 32 lanes, one point each (vol L2-resident: 67MB < 192MB) ----
    const long p = chunk + lane;
    if (p < N) {
      const float* xp = x + 4 * p;
      float dx = xp[0] + 5.f * (dd[0] + bf0);
      float dy = xp[1] + 5.f * (dd[1] + bf1);
      float dz = xp[2] + 5.f * (dd[2] + bf2);
      dx = fminf(fmaxf(dx, 0.f), 255.f);
      dy = fminf(fmaxf(dy, 0.f), 255.f);
      dz = fminf(fmaxf(dz, 0.f), 255.f);
      const float fx0 = floorf(dx), fy0 = floorf(dy), fz0 = floorf(dz);
      const int ix0 = (int)fx0, iy0 = (int)fy0, iz0 = (int)fz0;
      const int ix1 = min(ix0 + 1, 255), iy1 = min(iy0 + 1, 255), iz1 = min(iz0 + 1, 255);
      const float fx = dx - fx0, fy = dy - fy0, fz = dz - fz0;
      const float gx = 1.f - fx, gy = 1.f - fy, gz = 1.f - fz;
#define VOL(ix, iy, iz) vol[((long)(ix) * 256 + (iy)) * 256 + (iz)]
      const float r =
          VOL(ix0, iy0, iz0) * gx * gy * gz + VOL(ix1, iy0, iz0) * fx * gy * gz +
          VOL(ix0, iy1, iz0) * gx * fy * gz + VOL(ix0, iy0, iz1) * gx * gy * fz +
          VOL(ix1, iy1, iz0) * fx * fy * gz + VOL(ix1, iy0, iz1) * fx * gy * fz +
          VOL(ix0, iy1, iz1) * gx * fy * fz + VOL(ix1, iy1, iz1) * fx * fy * fz;
#undef VOL
      out[p] = r;
    }
  }
}

extern "C" void kernel_launch(void* const* d_in, const int* in_sizes, int n_in,
                              void* d_out, int out_size, void* d_ws, size_t ws_size,
                              hipStream_t stream) {
  (void)in_sizes; (void)n_in; (void)d_ws; (void)ws_size;
  const float* x   = (const float*)d_in[0];
  const float* W1  = (const float*)d_in[1];
  const float* b1  = (const float*)d_in[2];
  const float* W2  = (const float*)d_in[3];
  const float* b2  = (const float*)d_in[4];
  const float* W3  = (const float*)d_in[5];
  const float* b3  = (const float*)d_in[6];
  const float* Wf  = (const float*)d_in[7];
  const float* bf  = (const float*)d_in[8];
  const float* vol = (const float*)d_in[9];
  float* out = (float*)d_out;

  const long N = (long)out_size;                 // 2,097,152 points
  const int grid = (int)((N + PPB - 1) / PPB);   // 1024 blocks of 256 threads
  deform_sample_kernel<<<grid, 256, 0, stream>>>(x, W1, b1, W2, b2, W3, b3, Wf, bf, vol, out, N);
}
